// LogStorageDiagonalCell_29497835389003
// MI455X (gfx1250) — matrix operand order repaired and checked by reading, not compiled, not run on hardware
//
#include <hip/hip_runtime.h>
#include <hip/hip_bf16.h>

// ---------------------------------------------------------------------------
// Types
// ---------------------------------------------------------------------------
typedef __attribute__((ext_vector_type(16))) __bf16 v16bf;
typedef __attribute__((ext_vector_type(8)))  __bf16 v8bf;
typedef __attribute__((ext_vector_type(8)))  float  v8f;
typedef __attribute__((ext_vector_type(4)))  unsigned int u32x4;
typedef __attribute__((ext_vector_type(8)))  int i32x8;
typedef __attribute__((ext_vector_type(4)))  int i32x4;

#define BM 128
#define BN 128
#define BK 32
#define LDT 40      // padded LDS row stride in bf16 elems (80B = 64B data + 16B pad)

#if defined(__has_builtin)
#if __has_builtin(__builtin_amdgcn_tensor_load_to_lds) && \
    __has_builtin(__builtin_amdgcn_s_wait_tensorcnt)
#define USE_TDM 1
#endif
#endif

// ---------------------------------------------------------------------------
// Fallback: CDNA5 async global->LDS copy (tracked by ASYNCcnt)
// ---------------------------------------------------------------------------
__device__ __forceinline__ void async_g2l_b128(unsigned lds_addr, unsigned goff,
                                               unsigned long long base) {
    asm volatile("global_load_async_to_lds_b128 %0, %1, %2"
                 :: "v"(lds_addr), "v"(goff), "s"(base)
                 : "memory");
}
__device__ __forceinline__ void wait_asynccnt0() {
    asm volatile("s_wait_asynccnt 0" ::: "memory");
}

// ---------------------------------------------------------------------------
// f32 -> bf16 conversion (grid-stride)
// ---------------------------------------------------------------------------
__global__ void f32_to_bf16_kernel(const float* __restrict__ in,
                                   __bf16* __restrict__ out, size_t n) {
    size_t i = blockIdx.x * (size_t)blockDim.x + threadIdx.x;
    size_t stride = gridDim.x * (size_t)blockDim.x;
    for (; i < n; i += stride) out[i] = (__bf16)in[i];
}

// ---------------------------------------------------------------------------
// Tiled bf16 GEMM:  Out[m,n] = act( sum_k A[m,k]*W[n,k] + bias[n] )
//   A: [M,K] row-major bf16 (x), W: [N,K] row-major bf16 (weights, K-contig)
//   mode==0 : identity (xw + b), mode==1 : sigmoid (delta)
// Block = 256 threads = 8 waves (2x4); wave tile 64x32 (4x2 WMMA tiles).
// Tile staging: TDM tensor_load_to_lds (wave 0 issues descriptors; the D#
// padding fields reproduce the LDT=40 padded row stride), falling back to
// per-lane GLOBAL_LOAD_ASYNC_TO_LDS_B128 if the builtin is unavailable.
// K-loop is unrolled by 2 so both LDS buffer addresses are compile-time.
// Requires K % BK == 0; KT = K/BK may be odd (K=1024 -> KT=32).
// ---------------------------------------------------------------------------
__global__ __launch_bounds__(256)
void dual_gemm_bf16_kernel(const __bf16* __restrict__ A,
                           const __bf16* __restrict__ W,
                           const float* __restrict__ bias,
                           float* __restrict__ Out,
                           int M, int N, int K, int mode) {
    __shared__ __bf16 sA[2][BM * LDT];
    __shared__ __bf16 sB[2][BN * LDT];

    const int tid  = threadIdx.x;
    const int lane = tid & 31;
    const int wave = tid >> 5;
    const int wm   = wave >> 2;     // 0..1 : wave row
    const int wn   = wave & 3;      // 0..3 : wave col
    const int half = lane >> 4;     // 0/1  : half-wave
    const int l16  = lane & 15;

    const int n0 = blockIdx.x * BN;
    const int m0 = blockIdx.y * BM;

    const unsigned long long baseA = (unsigned long long)(uintptr_t)A;
    const unsigned long long baseW = (unsigned long long)(uintptr_t)W;
    const int KT = K / BK;

#ifdef USE_TDM
    // ---- TDM descriptor staging (wave 0 only; EXEC-independent DMA) ----
    // D# group1 (shared invariants): data_size=2B, pad_enable,
    // pad_interval=3 (16 DWORDs = 64B of data), pad_amount=3 (4 DWORDs = 16B)
    // -> LDS row stride 80B == LDT*2.  tile = 32 (K) x 128 (rows).
    auto tdm_issue = [&](unsigned lds_base, unsigned long long gaddr,
                         int rows_total) {
        u32x4 g0;
        g0[0] = 1u;                                        // count=1 (valid)
        g0[1] = lds_base;                                  // lds_addr
        g0[2] = (unsigned)(gaddr & 0xffffffffull);         // global_addr[31:0]
        g0[3] = ((unsigned)(gaddr >> 32) & 0x01ffffffu)    // global_addr[56:32]
              | (2u << 30);                                // type=2 ("image")
        i32x8 g1;
        g1[0] = (1 << 16)          // data_size = 1 -> 2 bytes
              | (1 << 20)          // pad_enable
              | (3 << 22)          // pad_interval: 16 DWORDs
              | (3 << 25);         // pad_amount: 4 DWORDs
        g1[1] = (int)((K & 0xffff) << 16);                         // tensor_dim0[15:0]
        g1[2] = (int)(((K >> 16) & 0xffff)                         // tensor_dim0[31:16]
              | ((rows_total & 0xffff) << 16));                    // tensor_dim1[15:0]
        g1[3] = (int)(((rows_total >> 16) & 0xffff)                // tensor_dim1[31:16]
              | (BK << 16));                                       // tile_dim0 = 32
        g1[4] = BM;                                                // tile_dim1 = 128
        g1[5] = K;                                                 // dim0_stride[31:0]
        g1[6] = 0;                                                 // stride hi / dim1_stride lo
        g1[7] = 0;
        i32x4 gz4 = {0, 0, 0, 0};
        i32x8 gz8 = {0, 0, 0, 0, 0, 0, 0, 0};
        // 6-arg form (amdgpu-toolchain / clang-23 therock headers)
        __builtin_amdgcn_tensor_load_to_lds(g0, g1, gz4, gz4, gz8, 0);
    };
    const unsigned sAbase[2] = { (unsigned)(uintptr_t)&sA[0][0],
                                 (unsigned)(uintptr_t)&sA[1][0] };
    const unsigned sBbase[2] = { (unsigned)(uintptr_t)&sB[0][0],
                                 (unsigned)(uintptr_t)&sB[1][0] };

    auto stage = [&](int buf, int kt) {
        if (wave == 0) {
            const unsigned long long kb = (unsigned long long)kt * BK * sizeof(__bf16);
            tdm_issue(sAbase[buf], baseA + (unsigned long long)m0 * K * 2 + kb, M);
            tdm_issue(sBbase[buf], baseW + (unsigned long long)n0 * K * 2 + kb, N);
        }
    };
    auto finish = [&]() {
        if (wave == 0) __builtin_amdgcn_s_wait_tensorcnt(0);
        __syncthreads();
    };
#else
    // ---- per-lane async staging: 2 x 16B chunks per tile per thread ----
    int row[2], col[2];
#pragma unroll
    for (int c = 0; c < 2; ++c) {
        int e = (c * 256 + tid) * 8;   // over 128x32 tile
        row[c] = e >> 5;
        col[c] = e & 31;
    }
    unsigned goffA[2], goffW[2], ldsA[2][2], ldsB[2][2];
#pragma unroll
    for (int c = 0; c < 2; ++c) {
        goffA[c] = (unsigned)(((size_t)(m0 + row[c]) * K + col[c]) * sizeof(__bf16));
        goffW[c] = (unsigned)(((size_t)(n0 + row[c]) * K + col[c]) * sizeof(__bf16));
#pragma unroll
        for (int bb = 0; bb < 2; ++bb) {
            ldsA[bb][c] = (unsigned)(uintptr_t)&sA[bb][row[c] * LDT + col[c]];
            ldsB[bb][c] = (unsigned)(uintptr_t)&sB[bb][row[c] * LDT + col[c]];
        }
    }
    auto stage = [&](int buf, int kt) {
        const unsigned kb = (unsigned)(kt * BK * sizeof(__bf16));
#pragma unroll
        for (int c = 0; c < 2; ++c) {
            async_g2l_b128(ldsA[buf][c], goffA[c] + kb, baseA);
            async_g2l_b128(ldsB[buf][c], goffW[c] + kb, baseW);
        }
    };
    auto finish = [&]() {
        wait_asynccnt0();
        __syncthreads();
    };
#endif

    v8f acc[4][2];
#pragma unroll
    for (int i = 0; i < 4; ++i)
#pragma unroll
        for (int j = 0; j < 2; ++j) acc[i][j] = (v8f){0.f,0.f,0.f,0.f,0.f,0.f,0.f,0.f};

    // One K-step of fragment loads + 8 WMMAs on a compile-time buffer index.
    auto compute = [&](int cur) {
        // A 16x32 bf16 frag: lane<16 -> M=l16, K={0..7,16..23}; lane>=16 -> +8
        v16bf af[4];
#pragma unroll
        for (int i = 0; i < 4; ++i) {
            const int m = wm * 64 + i * 16 + l16;
            union { v16bf v; v8bf h[2]; } u;
            u.h[0] = *(const v8bf*)&sA[cur][m * LDT + half * 8];
            u.h[1] = *(const v8bf*)&sA[cur][m * LDT + half * 8 + 16];
            af[i] = u.v;
        }
        // B 32x16 bf16 frag: lane<16 -> N=l16, K={0..7,16..23}; lane>=16 -> +8
        v16bf bfr[2];
#pragma unroll
        for (int j = 0; j < 2; ++j) {
            const int n = wn * 32 + j * 16 + l16;
            union { v16bf v; v8bf h[2]; } u;
            u.h[0] = *(const v8bf*)&sB[cur][n * LDT + half * 8];
            u.h[1] = *(const v8bf*)&sB[cur][n * LDT + half * 8 + 16];
            bfr[j] = u.v;
        }
#pragma unroll
        for (int i = 0; i < 4; ++i)
#pragma unroll
            for (int j = 0; j < 2; ++j)
                acc[i][j] = __builtin_amdgcn_wmma_f32_16x16x32_bf16(
                    false, af[i], false, bfr[j], (short)0, acc[i][j], false, false);
    };

    // ---- prologue ----
    stage(0, 0);
    finish();

    // ---- main loop, unrolled x2 so buffer indices are literals ----
    for (int kt = 0; kt < KT; kt += 2) {
        if (kt + 1 < KT) stage(1, kt + 1);
        compute(0);
        finish();
        if (kt + 1 >= KT) break;
        if (kt + 2 < KT) stage(0, kt + 2);
        compute(1);
        finish();
    }

    // ---- epilogue: C/D layout -> lanes 0-15: M=v, lanes 16-31: M=v+8; N=l16 ----
#pragma unroll
    for (int j = 0; j < 2; ++j) {
        const int n = n0 + wn * 32 + j * 16 + l16;
        const float bj = bias[n];
#pragma unroll
        for (int i = 0; i < 4; ++i) {
            const int mb = m0 + wm * 64 + i * 16 + half * 8;
#pragma unroll
            for (int v = 0; v < 8; ++v) {
                float val = acc[i][j][v] + bj;
                if (mode) val = 1.f / (1.f + __expf(-val));
                Out[(size_t)(mb + v) * N + n] = val;
            }
        }
    }
}

// ---------------------------------------------------------------------------
// Sequential scan over T. One thread per (b,d) channel; h carried linearly,
// log/sign computed only for output.
// ---------------------------------------------------------------------------
__global__ __launch_bounds__(256)
void scan_kernel(const float* __restrict__ xw, const float* __restrict__ delta,
                 const float* __restrict__ x,
                 const float* __restrict__ log_h0, const float* __restrict__ sign_h0,
                 const float* __restrict__ r_h, const float* __restrict__ b_gate,
                 float* __restrict__ out_output, float* __restrict__ out_logh,
                 float* __restrict__ out_signh,  float* __restrict__ out_hlin,
                 int T, int BD, int D) {
    const int gid = blockIdx.x * blockDim.x + threadIdx.x;
    if (gid >= BD) return;
    const int d = gid % D;

    const float r  = r_h[d];
    const float bg = b_gate[d];

    const float lh0 = log_h0[gid];
    const float sh0 = sign_h0[gid];
    out_logh[gid]  = lh0;       // row t=0 of [T+1,B,D]
    out_signh[gid] = sh0;

    float h = sh0 * __expf(lh0);

    size_t off = gid;
    for (int t = 0; t < T; ++t, off += BD) {
        if (t + 1 < T) {
            __builtin_prefetch(xw    + off + BD, 0, 0);
            __builtin_prefetch(delta + off + BD, 0, 0);
            __builtin_prefetch(x     + off + BD, 0, 0);
        }
        const float xwv = xw[off];
        const float dl  = delta[off];
        const float xv  = x[off];

        const float v    = xwv + r * h;
        const float cand = tanhf(v);
        h = (1.f - dl) * h + dl * cand;

        const float lg = __logf(fmaxf(fabsf(h), 1e-30f));
        const float sg = (h >= 0.f) ? 1.f : -1.f;

        const float gi   = h + xv + bg;
        const float gate = gi / (1.f + __expf(-gi));   // silu(gi)

        out_output[off]      = h * gate;
        out_hlin[off]        = h;
        out_logh[off + BD]   = lg;   // rows 1..T
        out_signh[off + BD]  = sg;
    }
}

// ---------------------------------------------------------------------------
// Host-side launcher
// ---------------------------------------------------------------------------
static inline size_t align_up(size_t v, size_t a) { return (v + a - 1) & ~(a - 1); }

extern "C" void kernel_launch(void* const* d_in, const int* in_sizes, int n_in,
                              void* d_out, int out_size, void* d_ws, size_t ws_size,
                              hipStream_t stream) {
    const float* x       = (const float*)d_in[0];
    const float* log_h0  = (const float*)d_in[1];
    const float* sign_h0 = (const float*)d_in[2];
    const float* W_x     = (const float*)d_in[3];
    const float* r_h     = (const float*)d_in[4];
    const float* W_delta = (const float*)d_in[5];
    const float* b       = (const float*)d_in[6];
    const float* b_delta = (const float*)d_in[7];
    const float* b_gate  = (const float*)d_in[8];

    const int D   = in_sizes[4];           // r_h is [D]
    const int BD  = in_sizes[1];           // log_h0 is [B,D]
    const int B   = BD / D;
    const int TBD = in_sizes[0];           // x is [T,B,D]
    const int T   = TBD / BD;
    const int M   = T * B;                 // GEMM rows
    const int N   = D;
    const int K   = D;

    // ---- workspace carve-up ----
    char* ws = (char*)d_ws;
    size_t off = 0;
    __bf16* x_bf  = (__bf16*)(ws + off); off = align_up(off + sizeof(__bf16) * (size_t)TBD, 256);
    __bf16* wx_bf = (__bf16*)(ws + off); off = align_up(off + sizeof(__bf16) * (size_t)D * D, 256);
    __bf16* wd_bf = (__bf16*)(ws + off); off = align_up(off + sizeof(__bf16) * (size_t)D * D, 256);
    float*  xw    = (float*)(ws + off);  off = align_up(off + sizeof(float) * (size_t)TBD, 256);
    float*  delta = (float*)(ws + off);  off = align_up(off + sizeof(float) * (size_t)TBD, 256);

    // ---- output carve-up: (output, log_h, sign_h, h_linear) ----
    const size_t S1 = (size_t)TBD;            // [T,B,D]
    const size_t S2 = (size_t)(T + 1) * BD;   // [T+1,B,D]
    float* out_output = (float*)d_out;
    float* out_logh   = out_output + S1;
    float* out_signh  = out_logh + S2;
    float* out_hlin   = out_signh + S2;

    // ---- 1) conversions to bf16 ----
    f32_to_bf16_kernel<<<2048, 256, 0, stream>>>(x, x_bf, (size_t)TBD);
    f32_to_bf16_kernel<<<512, 256, 0, stream>>>(W_x, wx_bf, (size_t)D * D);
    f32_to_bf16_kernel<<<512, 256, 0, stream>>>(W_delta, wd_bf, (size_t)D * D);

    // ---- 2) the two GEMMs ----
    dim3 ggrid(N / BN, M / BM);
    dual_gemm_bf16_kernel<<<ggrid, 256, 0, stream>>>(x_bf, wx_bf, b,       xw,    M, N, K, 0);
    dual_gemm_bf16_kernel<<<ggrid, 256, 0, stream>>>(x_bf, wd_bf, b_delta, delta, M, N, K, 1);

    // ---- 3) sequential scan ----
    scan_kernel<<<(BD + 255) / 256, 256, 0, stream>>>(
        xw, delta, x, log_h0, sign_h0, r_h, b_gate,
        out_output, out_logh, out_signh, out_hlin, T, BD, D);
}
